// NaiveCustomLSTM_31997506356094
// MI455X (gfx1250) — matrix operand order repaired and checked by reading, not compiled, not run on hardware
//
#include <hip/hip_runtime.h>
#include <hip/hip_bf16.h>
#include <cstdint>

// ---------------------------------------------------------------------------
// LSTM  (B=64, T=2048, I=H=256) for MI455X / gfx1250 (wave32, WMMA bf16)
//
// Phase 1: preact[t][b][4H] = x @ [U_i|U_f|U_c|U_o] + b   (parallel GEMM, WMMA)
// Phase 2: persistent single-workgroup scan over T steps:
//          gates = act(preact_t + h @ [V_i|V_f|V_c|V_o]) (WMMA, h in LDS,
//          c in registers, preact DMA'd to LDS with async-to-LDS loads).
//
// Workspace layout (bytes), assumed ws_size >= ~337 MB:
//   [0,            67108864)   packedA : x as bf16 in WMMA A-fragment layout
//   [67108864,     67633152)   packedU : U concat bf16, B-fragment layout
//   [67633152,     68157440)   packedV : V concat bf16, B-fragment layout
//   [68157440,    336592896)   preact  : bf16 [T][B][4H]
// ---------------------------------------------------------------------------

#define LSTM_B   64
#define LSTM_T   2048
#define LSTM_H   256
#define G4H      1024   // 4 gates * H

typedef __attribute__((ext_vector_type(16))) __bf16 bf16x16;
typedef __attribute__((ext_vector_type(8)))  __bf16 bf16x8;
typedef __attribute__((ext_vector_type(8)))  float  f32x8;

__device__ __forceinline__ unsigned short f2bf(float f) {
    unsigned int u = __float_as_uint(f);
    u = (u + 0x7FFFu + ((u >> 16) & 1u)) >> 16;   // round-to-nearest-even
    return (unsigned short)u;
}
__device__ __forceinline__ float bf2f(unsigned short s) {
    return __uint_as_float(((unsigned int)s) << 16);
}

// ---------------------------------------------------------------------------
// Pack x[B*T, 256] (f32 row-major) -> A-fragment-major bf16.
// A 16x32 bf16 fragment (ISA 7.12.2): lane L holds row M=(L&15),
// K = kbase..kbase+7 and kbase+16..kbase+23, kbase=(L>>4)*8.
// packedA[((mt*8+kt)*32+lane)*16 + j],  k = kt*32 + kbase + (j<8 ? j : j+8)
// ---------------------------------------------------------------------------
__global__ __launch_bounds__(256) void k_pack_x(const float* __restrict__ x,
                                                unsigned short* __restrict__ pA) {
    int idx  = blockIdx.x * 256 + threadIdx.x;     // 33,554,432 elements
    int j    = idx & 15;
    int lane = (idx >> 4) & 31;
    int kt   = (idx >> 9) & 7;
    int mt   = idx >> 12;
    int m    = mt * 16 + (lane & 15);
    int kbase = (lane >> 4) * 8;
    int k    = kt * 32 + kbase + (j < 8 ? j : j + 8);
    pA[idx] = f2bf(x[(size_t)m * 256 + k]);
}

// ---------------------------------------------------------------------------
// Pack 4 gate weights [256,256] f32 -> concat [256,1024] B-fragment bf16.
// B 32x16 bf16 fragment: lane L holds column N=(L&15), K=(L>>4)*8 + (j<8 ? j : j+8).
// packedB[((kt*64+nt)*32+lane)*16 + j]
// ---------------------------------------------------------------------------
__global__ __launch_bounds__(256) void k_pack_w(const float* __restrict__ W0,
                                                const float* __restrict__ W1,
                                                const float* __restrict__ W2,
                                                const float* __restrict__ W3,
                                                unsigned short* __restrict__ pB) {
    int idx  = blockIdx.x * 256 + threadIdx.x;     // 262,144 elements
    int j    = idx & 15;
    int lane = (idx >> 4) & 31;
    int nt   = (idx >> 9) & 63;
    int kt   = idx >> 15;
    int n    = nt * 16 + (lane & 15);
    int k    = kt * 32 + (lane >> 4) * 8 + (j < 8 ? j : j + 8);
    const float* W = (n < 256) ? W0 : (n < 512) ? W1 : (n < 768) ? W2 : W3;
    pB[idx] = f2bf(W[(size_t)k * 256 + (n & 255)]);
}

// ---------------------------------------------------------------------------
// Phase-1 GEMM: preact[t][b][n] = x @ Ucat + bias  (bf16 out)
// Block = 256 threads = 8 waves. Wave w -> mtile = blockIdx.x*8+w (16 rows),
// 4 ntiles = blockIdx.y*4.. (64 cols). K loop = 8 x (K=32) WMMA steps.
// ---------------------------------------------------------------------------
__global__ __launch_bounds__(256) void k_gemm_preact(
        const unsigned short* __restrict__ pA,
        const unsigned short* __restrict__ pU,
        const float* __restrict__ bi, const float* __restrict__ bf_,
        const float* __restrict__ bc, const float* __restrict__ bo,
        unsigned short* __restrict__ preact) {
    int w     = threadIdx.x >> 5;
    int lane  = threadIdx.x & 31;
    int mtile = blockIdx.x * 8 + w;
    int ntb   = blockIdx.y * 4;

    f32x8 acc[4] = {};
    for (int kt = 0; kt < 8; ++kt) {
        bf16x16 a = *(const bf16x16*)(pA + ((((size_t)mtile * 8 + kt) * 32 + lane) << 4));
#pragma unroll
        for (int nt = 0; nt < 4; ++nt) {
            bf16x16 b = *(const bf16x16*)(pU + ((((size_t)kt * 64 + (ntb + nt)) * 32 + lane) << 4));
            acc[nt] = __builtin_amdgcn_wmma_f32_16x16x32_bf16(
                false, a, false, b, (short)0, acc[nt], false, false);
        }
    }

    // C fragment (ISA 7.12.2): VGPR r -> M = r + (lane>=16 ? 8:0), N = lane&15
    int rbase = mtile * 16 + ((lane >> 4) << 3);
#pragma unroll
    for (int nt = 0; nt < 4; ++nt) {
        int n = (ntb + nt) * 16 + (lane & 15);          // 0..1023
        const float* bp = (n < 256) ? bi : (n < 512) ? bf_ : (n < 768) ? bc : bo;
        float bias = bp[n & 255];
#pragma unroll
        for (int r = 0; r < 8; ++r) {
            int m = rbase + r;                          // row in [0, B*T)
            int b = m >> 11;                            // m / T
            int t = m & (LSTM_T - 1);                   // m % T
            preact[((size_t)t * LSTM_B + b) * G4H + n] = f2bf(acc[nt][r] + bias);
        }
    }
}

// ---------------------------------------------------------------------------
// Phase-2: persistent single-workgroup scan. 512 threads = 16 waves.
// Wave w owns columns [w*64, w*64+64)  -> gate = w>>2 (i,f,g,o).
// Per step: 4 mtiles x 4 ntiles x 8 k-steps = 128 WMMAs per wave, while the
// step's 128KB preact block is DMA'd into LDS with async-to-LDS loads.
// LDS: h bf16 (32KB) | 4 gate bufs bf16 (128KB) | preact stage (128KB) = 288KB
// Cell state c lives in 32 registers per thread (static element mapping).
// ---------------------------------------------------------------------------
#define LDS_H_OFF    0
#define LDS_G_OFF    (32 * 1024)
#define LDS_PRE_OFF  (160 * 1024)

__global__ __launch_bounds__(512) void k_lstm_scan(
        const unsigned short* __restrict__ pV,
        const unsigned short* __restrict__ preact,
        float* __restrict__ outH, float* __restrict__ outHT,
        float* __restrict__ outCT) {
    extern __shared__ char smem[];
    unsigned short* h_bf   = (unsigned short*)(smem + LDS_H_OFF);   // [64][256]
    unsigned short* gbuf   = (unsigned short*)(smem + LDS_G_OFF);   // [4][64][256]
    unsigned short* pre_ls = (unsigned short*)(smem + LDS_PRE_OFF); // [64][1024]

    const int tid  = threadIdx.x;
    const int w    = tid >> 5;
    const int lane = tid & 31;
    const int gate = w >> 2;
    const int ntb  = w * 4;
    const int halfsel = (lane >> 4) * 8;    // A-frag k-base / C-frag row-base
    const int colsel  = lane & 15;

    // LDS byte offsets for the async-to-LDS DMA (low 32 bits of a shared
    // address are the workgroup-relative LDS byte offset).
    const unsigned int ldsPreOff = (unsigned int)(uintptr_t)pre_ls + (unsigned int)tid * 16u;
    const unsigned int gOff      = (unsigned int)tid * 16u;

    float c_reg[32];
#pragma unroll
    for (int i = 0; i < 32; ++i) c_reg[i] = 0.0f;
    for (int e = tid; e < LSTM_B * LSTM_H; e += 512) h_bf[e] = 0;
    __syncthreads();

#pragma unroll 1
    for (int t = 0; t < LSTM_T; ++t) {
        // ---- kick off async DMA of this step's preact block into LDS ----
        // Each thread copies 256B as 16 x b128; ISA adds INST_OFFSET to both
        // the LDS and the global address, so one (vdst,vaddr) pair suffices.
        {
            unsigned long long gbase =
                (unsigned long long)(uintptr_t)(preact + (size_t)t * LSTM_B * G4H);
            asm volatile(
                "global_load_async_to_lds_b128 %0, %1, %2 offset:0\n\t"
                "global_load_async_to_lds_b128 %0, %1, %2 offset:8192\n\t"
                "global_load_async_to_lds_b128 %0, %1, %2 offset:16384\n\t"
                "global_load_async_to_lds_b128 %0, %1, %2 offset:24576\n\t"
                "global_load_async_to_lds_b128 %0, %1, %2 offset:32768\n\t"
                "global_load_async_to_lds_b128 %0, %1, %2 offset:40960\n\t"
                "global_load_async_to_lds_b128 %0, %1, %2 offset:49152\n\t"
                "global_load_async_to_lds_b128 %0, %1, %2 offset:57344\n\t"
                "global_load_async_to_lds_b128 %0, %1, %2 offset:65536\n\t"
                "global_load_async_to_lds_b128 %0, %1, %2 offset:73728\n\t"
                "global_load_async_to_lds_b128 %0, %1, %2 offset:81920\n\t"
                "global_load_async_to_lds_b128 %0, %1, %2 offset:90112\n\t"
                "global_load_async_to_lds_b128 %0, %1, %2 offset:98304\n\t"
                "global_load_async_to_lds_b128 %0, %1, %2 offset:106496\n\t"
                "global_load_async_to_lds_b128 %0, %1, %2 offset:114688\n\t"
                "global_load_async_to_lds_b128 %0, %1, %2 offset:122880"
                :: "v"(ldsPreOff), "v"(gOff), "s"(gbase) : "memory");
        }
        if (t + 1 < LSTM_T)   // warm L2 for next step's preactivations
            __builtin_prefetch(preact + (size_t)(t + 1) * LSTM_B * G4H +
                               (size_t)tid * 128, 0, 0);

        // ---- recurrent GEMM: h[64,256] @ Vcat[256,1024] (overlaps the DMA) ----
        f32x8 acc[4][4] = {};
#pragma unroll 1
        for (int kt = 0; kt < 8; ++kt) {
            bf16x16 afr[4];
#pragma unroll
            for (int mt = 0; mt < 4; ++mt) {
                int row = mt * 16 + colsel;
                int k0  = kt * 32 + halfsel;
                bf16x8 lo = *(const bf16x8*)(h_bf + row * 256 + k0);
                bf16x8 hi = *(const bf16x8*)(h_bf + row * 256 + k0 + 16);
                bf16x16 a;
#pragma unroll
                for (int j = 0; j < 8; ++j) { a[j] = lo[j]; a[8 + j] = hi[j]; }
                afr[mt] = a;
            }
#pragma unroll
            for (int nt = 0; nt < 4; ++nt) {
                bf16x16 b = *(const bf16x16*)(pV + ((((size_t)kt * 64 + ntb + nt) * 32 + lane) << 4));
#pragma unroll
                for (int mt = 0; mt < 4; ++mt)
                    acc[mt][nt] = __builtin_amdgcn_wmma_f32_16x16x32_bf16(
                        false, afr[mt], false, b, (short)0, acc[mt][nt], false, false);
            }
        }

        // ---- wait for the DMA, make it visible workgroup-wide ----
        asm volatile("s_wait_asynccnt 0" ::: "memory");
        __syncthreads();

        // ---- epilogue: + x-preactivation (from LDS), activation, gate buf ----
#pragma unroll
        for (int mt = 0; mt < 4; ++mt) {
#pragma unroll
            for (int nt = 0; nt < 4; ++nt) {
                int n    = (ntb + nt) * 16 + colsel;    // global column 0..1023
                int gcol = n & 255;                     // column within gate
#pragma unroll
                for (int r = 0; r < 8; ++r) {
                    int b = mt * 16 + halfsel + r;      // batch row 0..63
                    float v = acc[mt][nt][r] + bf2f(pre_ls[b * G4H + n]);
                    float g = (gate == 2) ? tanhf(v) : (1.0f / (1.0f + __expf(-v)));
                    gbuf[gate * (LSTM_B * LSTM_H) + b * LSTM_H + gcol] = f2bf(g);
                }
            }
        }
        __syncthreads();

        // ---- combine gates -> c (registers), h (LDS); emit h_seq (fp32) ----
#pragma unroll
        for (int i = 0; i < 32; ++i) {
            int e   = tid + i * 512;
            int b   = e >> 8;
            int col = e & 255;
            float iv = bf2f(gbuf[0 * 16384 + e]);
            float fv = bf2f(gbuf[1 * 16384 + e]);
            float gv = bf2f(gbuf[2 * 16384 + e]);
            float ov = bf2f(gbuf[3 * 16384 + e]);
            float c_new = fv * c_reg[i] + iv * gv;
            float h_new = ov * tanhf(c_new);
            c_reg[i] = c_new;
            h_bf[e]  = f2bf(h_new);
            outH[((size_t)b * LSTM_T + t) * LSTM_H + col] = h_new;
            if (t == LSTM_T - 1) { outHT[e] = h_new; outCT[e] = c_new; }
        }
        __syncthreads();
    }
}

// ---------------------------------------------------------------------------
extern "C" void kernel_launch(void* const* d_in, const int* in_sizes, int n_in,
                              void* d_out, int out_size, void* d_ws, size_t ws_size,
                              hipStream_t stream) {
    const float* x   = (const float*)d_in[0];
    const float* U_i = (const float*)d_in[1];
    const float* V_i = (const float*)d_in[2];
    const float* b_i = (const float*)d_in[3];
    const float* U_f = (const float*)d_in[4];
    const float* V_f = (const float*)d_in[5];
    const float* b_f = (const float*)d_in[6];
    const float* U_c = (const float*)d_in[7];
    const float* V_c = (const float*)d_in[8];
    const float* b_c = (const float*)d_in[9];
    const float* U_o = (const float*)d_in[10];
    const float* V_o = (const float*)d_in[11];
    const float* b_o = (const float*)d_in[12];

    char* ws = (char*)d_ws;
    unsigned short* pA  = (unsigned short*)(ws);
    unsigned short* pU  = (unsigned short*)(ws + 67108864);
    unsigned short* pV  = (unsigned short*)(ws + 67633152);
    unsigned short* pre = (unsigned short*)(ws + 68157440);

    k_pack_x<<<131072, 256, 0, stream>>>(x, pA);
    k_pack_w<<<1024, 256, 0, stream>>>(U_i, U_f, U_c, U_o, pU);
    k_pack_w<<<1024, 256, 0, stream>>>(V_i, V_f, V_c, V_o, pV);
    k_gemm_preact<<<dim3(1024, 16), 256, 0, stream>>>(pA, pU, b_i, b_f, b_c, b_o, pre);

    float* outH  = (float*)d_out;
    float* outHT = outH + (size_t)LSTM_B * LSTM_T * LSTM_H;
    float* outCT = outHT + (size_t)LSTM_B * LSTM_H;
    k_lstm_scan<<<1, 512, 288 * 1024, stream>>>(pV, pre, outH, outHT, outCT);
}
